// MultilayerGRU_10247791969060
// MI455X (gfx1250) — matrix operand order repaired and checked by reading, not compiled, not run on hardware
//
#include <hip/hip_runtime.h>

typedef __bf16 bf16;
typedef __attribute__((ext_vector_type(16))) __bf16        v16bf;
typedef __attribute__((ext_vector_type(8)))  float         v8f;
typedef __attribute__((ext_vector_type(8)))  unsigned int  u32x8;
typedef __attribute__((ext_vector_type(4)))  unsigned int  u32x4;

constexpr int kB = 64, kS = 512, kIN = 256, kH = 1024, kOUT = 256;
constexpr int kNWG = 64, kTPB = 256;                 // 64 WGs x 8 wave32 = 512 waves
constexpr int KT0  = (kIN + kH) / 32;                // 40 k-tiles, layer 0 (concat [x|h])
constexpr int KTI  = (2 * kH) / 32;                  // 64 k-tiles, inner layers (concat [cur|h])
constexpr int NT_H = kH / 16;                        // 64 n-tiles for H-wide gates
constexpr int KT_Y = kH / 32;                        // 32 k-tiles for output proj
constexpr int NT_Y = kOUT / 16;                      // 16 n-tiles for output proj

// ---- workspace layout (bytes) ----
constexpr size_t HDR_OFF = 0;                                  // barrier cnt/gen
constexpr size_t XBF_OFF = 256;
constexpr size_t XBF_SZ  = (size_t)kB * kS * kIN * 2;          // x in bf16
constexpr size_t P0_SZ   = (size_t)KT0 * NT_H * 512 * 2;       // packed layer-0 gate
constexpr size_t PZ0_OFF = XBF_OFF + XBF_SZ;
constexpr size_t PR0_OFF = PZ0_OFF + P0_SZ;
constexpr size_t PG0_OFF = PR0_OFF + P0_SZ;
constexpr size_t PI_SZ   = (size_t)KTI * NT_H * 512 * 2;       // packed inner gate
constexpr size_t PZ1_OFF = PG0_OFF + P0_SZ;
constexpr size_t PR1_OFF = PZ1_OFF + PI_SZ;
constexpr size_t PZ2_OFF = PR1_OFF + PI_SZ;
constexpr size_t PR2_OFF = PZ2_OFF + PI_SZ;
constexpr size_t PWY_SZ  = (size_t)KT_Y * NT_Y * 512 * 2;      // packed Why
constexpr size_t PWY_OFF = PR2_OFF + PI_SZ;
constexpr size_t ST_SZ   = (size_t)3 * kB * kH * 2;            // hidden states, bf16
constexpr size_t ST_OFF  = PWY_OFF + PWY_SZ;
constexpr size_t ZB_SZ   = (size_t)kB * kH * 4;                // z gate, f32
constexpr size_t ZB_OFF  = ST_OFF + ST_SZ;
constexpr size_t RS_SZ   = (size_t)kB * kH * 2;                // r*state, bf16
constexpr size_t RS_OFF  = ZB_OFF + ZB_SZ;
constexpr size_t H2_OFF  = RS_OFF + RS_SZ;                     // layer-2 history [S][B][H] bf16

struct GruParams {
  const bf16*  xbf;
  const bf16*  pz[3];
  const bf16*  pr[3];
  const bf16*  pg[3];
  const float* bz[3];
  const float* br[3];
  const float* bg[3];
  const bf16*  pwy;
  const float* bhy;
  bf16*  st;       // [3][B][H]
  float* zb;       // [B][H]
  bf16*  rs;       // [B][H]
  bf16*  h2;       // [S][B][H]
  float* out;      // layer_output [B][S][OUT] ++ hidden_state [B][3][H]
  unsigned* cnt;
  unsigned* gen;
};

// ---------- helpers ----------
__device__ __forceinline__ float sigm(float x) { return 1.f / (1.f + __expf(-x)); }

// Branch-free K-region: A rows stepped 64B/tile, B tiles stepped 1KB/tile.
// a_lane already points at this lane's first 16B chunk (row + half*8);
// second chunk is always +32B. b_lane already includes the lane's 32B offset.
__device__ __forceinline__ v8f gemm_region(v8f acc, const bf16* a_lane,
                                           const u32x8* b_lane, int ntiles) {
#pragma unroll 4
  for (int kt = 0; kt < ntiles; ++kt) {
    u32x4 lo = *(const u32x4*)a_lane;
    u32x4 hi = *(const u32x4*)(a_lane + 16);
    u32x8 wa;
    wa[0] = lo[0]; wa[1] = lo[1]; wa[2] = lo[2]; wa[3] = lo[3];
    wa[4] = hi[0]; wa[5] = hi[1]; wa[6] = hi[2]; wa[7] = hi[3];
    v16bf av = __builtin_bit_cast(v16bf, wa);
    v16bf bv = __builtin_bit_cast(v16bf, b_lane[0]);
    acc = __builtin_amdgcn_wmma_f32_16x16x32_bf16(false, av, false, bv, (short)0, acc,
                                                  false, false);
    a_lane += 32;   // next 16x32 A tile along K
    b_lane += 32;   // next 1KB packed B tile
  }
  return acc;
}

// One 16x16 output tile over concat A = [in_row | st_row], K = (n0+n1)*32.
__device__ __forceinline__ v8f wmma_tile(const bf16* in_row, int n0,
                                         const bf16* st_row, int n1,
                                         const bf16* btiles, int lane) {
  v8f acc;
#pragma unroll
  for (int i = 0; i < 8; ++i) acc[i] = 0.f;
  const int half = lane >> 4;
  const u32x8* b_lane = (const u32x8*)btiles + lane;
  acc = gemm_region(acc, in_row + half * 8, b_lane, n0);
  acc = gemm_region(acc, st_row + half * 8, b_lane + (size_t)n0 * 32, n1);
  return acc;
}

__device__ __forceinline__ void grid_sync(unsigned* cnt, unsigned* gen) {
  __syncthreads();
  if (threadIdx.x == 0) {
    __threadfence();
    unsigned g = __hip_atomic_load(gen, __ATOMIC_RELAXED, __HIP_MEMORY_SCOPE_AGENT);
    unsigned a = __hip_atomic_fetch_add(cnt, 1u, __ATOMIC_ACQ_REL, __HIP_MEMORY_SCOPE_AGENT);
    if (a == (unsigned)(kNWG - 1)) {
      __hip_atomic_store(cnt, 0u, __ATOMIC_RELAXED, __HIP_MEMORY_SCOPE_AGENT);
      __hip_atomic_store(gen, g + 1u, __ATOMIC_RELEASE, __HIP_MEMORY_SCOPE_AGENT);
    } else {
      while (__hip_atomic_load(gen, __ATOMIC_ACQUIRE, __HIP_MEMORY_SCOPE_AGENT) == g)
        __builtin_amdgcn_s_sleep(8);
    }
    __threadfence();
  }
  __syncthreads();
}

// ---------- prep kernels ----------
__global__ void k_cvt_bf16(const float* __restrict__ src, bf16* __restrict__ dst, int n) {
  int i = blockIdx.x * blockDim.x + threadIdx.x;
  if (i < n) dst[i] = (bf16)src[i];
}

// Pack concat [Wx | Wh] (both stored [N][K_row]) into WMMA B-operand tiles [nt][kt][lane][16].
__global__ void k_pack_w(const float* __restrict__ wx, int ldx,
                         const float* __restrict__ wh, int ldh,
                         int kin, int ktiles, int ntiles, bf16* __restrict__ dst) {
  int tid = blockIdx.x * blockDim.x + threadIdx.x;
  int total = ktiles * ntiles * 512;
  if (tid >= total) return;
  int e = tid & 15;
  int l = (tid >> 4) & 31;
  int tile = tid >> 9;
  int kt = tile % ktiles;
  int nt = tile / ktiles;
  int k = kt * 32 + (l >> 4) * 8 + (e >> 3) * 16 + (e & 7);   // B lane map: half h holds K 8h..8h+7, 16+8h..16+8h+7
  int n = nt * 16 + (l & 15);
  float v = (k < kin) ? wx[(size_t)n * ldx + k] : wh[(size_t)n * ldh + (k - kin)];
  dst[tid] = (bf16)v;
}

// ---------- persistent GRU kernel ----------
__global__ __launch_bounds__(kTPB) void k_gru_persistent(GruParams p) {
  const int lane  = threadIdx.x & 31;
  const int w     = blockIdx.x * (kTPB >> 5) + (threadIdx.x >> 5);   // 0..511
  const int gtid  = blockIdx.x * kTPB + threadIdx.x;
  const int nth   = kNWG * kTPB;
  const int half  = lane >> 4;
  const int am    = lane & 15;     // A-operand row within tile
  const int n_loc = lane & 15;     // C-operand column within tile
  const int mbase = half * 8;      // C-operand row base within tile

  for (int t = 0; t < kS; ++t) {
#pragma unroll
    for (int layer = 0; layer < 3; ++layer) {
      // all of these are compile-time constants after layer-loop unroll
      const int kin_t = (layer == 0) ? (kIN / 32) : (kH / 32);
      const int Kt    = (layer == 0) ? KT0 : KTI;
      const size_t cur_ld = (layer == 0) ? (size_t)kS * kIN : (size_t)kH;
      const bf16* cur = (layer == 0) ? (p.xbf + (size_t)t * kIN)
                                     : (p.st + (size_t)(layer - 1) * kB * kH);
      bf16* stL = p.st + (size_t)layer * kB * kH;

      { // ---- stage A: z,r = sigmoid([cur|st] @ [Wx|Wh]^T + b); fused epilogue ----
        int gate = w >> 8;            // 0 = z, 1 = r
        int mt   = (w >> 6) & 3;
        int nt   = w & 63;
        const bf16* packed = gate ? p.pr[layer] : p.pz[layer];
        size_t mrow = (size_t)(mt * 16 + am);
        v8f acc = wmma_tile(cur + mrow * cur_ld, kin_t,
                            stL + mrow * kH, Kt - kin_t,
                            packed + (size_t)nt * Kt * 512, lane);
        const float* bias = gate ? p.br[layer] : p.bz[layer];
#pragma unroll
        for (int i = 0; i < 8; ++i) {
          int m = mt * 16 + mbase + i;
          int n = nt * 16 + n_loc;
          float v = sigm(acc[i] + bias[n]);
          if (gate == 0) p.zb[m * kH + n] = v;
          else           p.rs[m * kH + n] = (bf16)(v * (float)stL[m * kH + n]);
        }
      }
      grid_sync(p.cnt, p.gen);

      if (w < 256) { // ---- stage B: g = tanh([cur|r*st] @ Wg^T + bg); h update ----
        int mt = w >> 6;
        int nt = w & 63;
        size_t mrow = (size_t)(mt * 16 + am);
        v8f acc = wmma_tile(cur + mrow * cur_ld, kin_t,
                            p.rs + mrow * kH, Kt - kin_t,
                            p.pg[layer] + (size_t)nt * Kt * 512, lane);
        const float* bias = p.bg[layer];
#pragma unroll
        for (int i = 0; i < 8; ++i) {
          int m = mt * 16 + mbase + i;
          int n = nt * 16 + n_loc;
          float g = tanhf(acc[i] + bias[n]);
          float z = p.zb[m * kH + n];
          float h = z * (float)stL[m * kH + n] + (1.f - z) * g;
          stL[m * kH + n] = (bf16)h;
          if (layer == 2) p.h2[((size_t)t * kB + m) * kH + n] = (bf16)h;
        }
      }
      grid_sync(p.cnt, p.gen);
    }
  }

  // ---- batched output projection: [S*B, H] @ Why^T + bhy -> layer_output ----
  for (int job = w; job < (kS * kB / 16) * NT_Y; job += kNWG * 8) {
    int mt = job >> 4;
    int nt = job & 15;
    size_t mrow = (size_t)mt * 16 + am;
    const bf16* arow = p.h2 + mrow * kH;
    v8f acc = wmma_tile(arow, 0, arow, KT_Y,
                        p.pwy + (size_t)nt * KT_Y * 512, lane);
#pragma unroll
    for (int i = 0; i < 8; ++i) {
      int ar = mt * 16 + mbase + i;     // row in [t][b] order
      int tt = ar >> 6;
      int b  = ar & 63;
      int o  = nt * 16 + n_loc;
      p.out[((size_t)b * kS + tt) * kOUT + o] = acc[i] + p.bhy[o];
    }
  }

  // ---- hidden_state [B][3][H] ----
  for (int idx = gtid; idx < 3 * kB * kH; idx += nth) {
    int l = idx / (kB * kH);
    int rem = idx - l * (kB * kH);
    int b = rem >> 10;
    int n = rem & (kH - 1);
    p.out[(size_t)kB * kS * kOUT + ((size_t)b * 3 + l) * kH + n] = (float)p.st[idx];
  }
}

// ---------- host ----------
extern "C" void kernel_launch(void* const* d_in, const int* in_sizes, int n_in,
                              void* d_out, int out_size, void* d_ws, size_t ws_size,
                              hipStream_t stream) {
  (void)in_sizes; (void)n_in; (void)out_size; (void)ws_size;
  const float* x    = (const float*)d_in[0];
  const float* Wxz0 = (const float*)d_in[1];
  const float* Whz0 = (const float*)d_in[2];
  const float* bhz0 = (const float*)d_in[3];
  const float* Wxr0 = (const float*)d_in[4];
  const float* Whr0 = (const float*)d_in[5];
  const float* bhr0 = (const float*)d_in[6];
  const float* Wxg0 = (const float*)d_in[7];
  const float* Whg0 = (const float*)d_in[8];
  const float* bhg0 = (const float*)d_in[9];
  const float* Wxz  = (const float*)d_in[10];   // [2][H][H]
  const float* Whz  = (const float*)d_in[11];
  const float* bhz  = (const float*)d_in[12];   // [2][H]
  const float* Wxr  = (const float*)d_in[13];
  const float* Whr  = (const float*)d_in[14];
  const float* bhr  = (const float*)d_in[15];
  const float* Why  = (const float*)d_in[16];
  const float* bhy  = (const float*)d_in[17];

  char* ws = (char*)d_ws;

  // zero barrier vars and h0 (deterministic every call; ws is not re-poisoned by harness)
  hipMemsetAsync(ws + HDR_OFF, 0, 256, stream);
  hipMemsetAsync(ws + ST_OFF, 0, ST_SZ, stream);

  // convert x to bf16
  int nx = kB * kS * kIN;
  k_cvt_bf16<<<(nx + 255) / 256, 256, 0, stream>>>(x, (bf16*)(ws + XBF_OFF), nx);

  // pack weights into WMMA B-operand tiles
  auto packLaunch = [&](const float* wx, int ldx, const float* wh, int ldh, int kin,
                        int kt, int nt, size_t off) {
    int total = kt * nt * 512;
    k_pack_w<<<(total + 255) / 256, 256, 0, stream>>>(wx, ldx, wh, ldh, kin, kt, nt,
                                                      (bf16*)(ws + off));
  };
  const size_t HH = (size_t)kH * kH;
  packLaunch(Wxz0, kIN, Whz0, kH, kIN, KT0, NT_H, PZ0_OFF);
  packLaunch(Wxr0, kIN, Whr0, kH, kIN, KT0, NT_H, PR0_OFF);
  packLaunch(Wxg0, kIN, Whg0, kH, kIN, KT0, NT_H, PG0_OFF);
  packLaunch(Wxz,       kH, Whz,       kH, kH, KTI, NT_H, PZ1_OFF);
  packLaunch(Wxr,       kH, Whr,       kH, kH, KTI, NT_H, PR1_OFF);
  packLaunch(Wxz + HH,  kH, Whz + HH,  kH, kH, KTI, NT_H, PZ2_OFF);
  packLaunch(Wxr + HH,  kH, Whr + HH,  kH, kH, KTI, NT_H, PR2_OFF);
  packLaunch(nullptr, 0, Why, kH, 0, KT_Y, NT_Y, PWY_OFF);

  GruParams p;
  p.xbf   = (const bf16*)(ws + XBF_OFF);
  p.pz[0] = (const bf16*)(ws + PZ0_OFF);
  p.pz[1] = (const bf16*)(ws + PZ1_OFF);
  p.pz[2] = (const bf16*)(ws + PZ2_OFF);
  p.pr[0] = (const bf16*)(ws + PR0_OFF);
  p.pr[1] = (const bf16*)(ws + PR1_OFF);
  p.pr[2] = (const bf16*)(ws + PR2_OFF);
  p.pg[0] = (const bf16*)(ws + PG0_OFF);
  p.pg[1] = p.pz[1];                 // inner-layer g-gate reuses z weights (source bug)
  p.pg[2] = p.pz[2];
  p.bz[0] = bhz0; p.bz[1] = bhz;      p.bz[2] = bhz + kH;
  p.br[0] = bhr0; p.br[1] = bhr;      p.br[2] = bhr + kH;
  p.bg[0] = bhg0; p.bg[1] = bhz;      p.bg[2] = bhz + kH;   // g reuses bhz on inner layers
  p.pwy = (const bf16*)(ws + PWY_OFF);
  p.bhy = bhy;
  p.st  = (bf16*)(ws + ST_OFF);
  p.zb  = (float*)(ws + ZB_OFF);
  p.rs  = (bf16*)(ws + RS_OFF);
  p.h2  = (bf16*)(ws + H2_OFF);
  p.out = (float*)d_out;
  p.cnt = (unsigned*)(ws + HDR_OFF);
  p.gen = p.cnt + 1;

  k_gru_persistent<<<kNWG, kTPB, 0, stream>>>(p);
}
